// MultiHeadAttention_11287174054597
// MI455X (gfx1250) — matrix operand order repaired and checked by reading, not compiled, not run on hardware
//
#include <hip/hip_runtime.h>
#include <cstdint>

typedef __bf16 bf16_t;
typedef __attribute__((ext_vector_type(16))) __bf16 v16bf;
typedef __attribute__((ext_vector_type(8)))  __bf16 v8bf;
typedef __attribute__((ext_vector_type(8)))  float   v8f;

#define T_SEQ 2048
#define DIM   64
#define NH    12
#define NB    4
#define BH    (NB*NH)      // 48
#define ROWS  (NB*T_SEQ)   // 8192
#define HD    (NH*DIM)     // 768

__device__ __forceinline__ v8f wmma_bf16(v16bf a, v16bf b, v8f c) {
    // D = A(16x32) * B(32x16) + C, fp32 accumulate
    return __builtin_amdgcn_wmma_f32_16x16x32_bf16(false, a, false, b, (short)0, c, false, false);
}

__device__ __forceinline__ v8f zero8() {
    v8f z;
#pragma unroll
    for (int i = 0; i < 8; ++i) z[i] = 0.f;
    return z;
}

__device__ __forceinline__ v16bf make_frag(v8bf lo, v8bf hi) {
    v16bf r;
#pragma unroll
    for (int i = 0; i < 8; ++i) { r[i] = lo[i]; r[i + 8] = hi[i]; }
    return r;
}

// 16-bit A fragment (16x32, M x K). Lane L: m = L&15, g = L>>4.
// half e -> k = 32*kc + 8*g + (e&7) + 16*(e>>3)  => two contiguous 16B chunks.
__device__ __forceinline__ v16bf load_a_frag(const bf16_t* rowBase, int kc, int g) {
    const bf16_t* p = rowBase + 32 * kc + 8 * g;
    return make_frag(*(const v8bf*)p, *(const v8bf*)(p + 16));
}

// 16-bit B fragment (32x16, K x N). Lane L: n = n0 + (L&15), g = L>>4;
// half e -> k = 32*kc + 8*g + (e&7) + 16*(e>>3) (same order as A).
__device__ __forceinline__ v16bf load_b_frag(const bf16_t* mat, int pitch, int kc, int n0, int lane) {
    const bf16_t* p = mat + (size_t)(32 * kc + 8 * (lane >> 4)) * pitch + n0 + (lane & 15);
    v16bf r;
#pragma unroll
    for (int i = 0; i < 8; ++i) { r[i] = p[(size_t)i * pitch]; r[i + 8] = p[(size_t)(i + 16) * pitch]; }
    return r;
}

// CDNA5 async copy: 16 bytes global -> LDS, tracked by ASYNCcnt (no VGPR round trip).
// GV addressing mode: vdst = LDS byte offset, v[2] pair = 64-bit global address.
__device__ __forceinline__ void async_copy16(void* lds_ptr, const void* gptr) {
    uint32_t lds_off = (uint32_t)(uintptr_t)lds_ptr;   // low 32 bits = LDS offset
    asm volatile("global_load_async_to_lds_b128 %0, %1, off"
                 :: "v"(lds_off), "v"(gptr)
                 : "memory");
}

__device__ __forceinline__ void wait_async0() {
    asm volatile("s_wait_asynccnt 0x0" ::: "memory");
}

// ---------------------------------------------------------------- QKV GEMM
// (8192x64) @ (64x768) for Wq/Wk/Wv (grid.z selects), bf16 output.
__global__ __launch_bounds__(256) void qkv_kernel(const float* __restrict__ x,
                                                  const float* __restrict__ Wq,
                                                  const float* __restrict__ Wk,
                                                  const float* __restrict__ Wv,
                                                  bf16_t* __restrict__ Qb,
                                                  bf16_t* __restrict__ Kb,
                                                  bf16_t* __restrict__ Vb) {
    __shared__ __align__(16) bf16_t ldsX[128 * 64];
    __shared__ __align__(16) bf16_t ldsW[64 * 64];
    const int tid = threadIdx.x;
    const int lane = tid & 31, wave = tid >> 5;
    const int g = lane >> 4;
    const int m0 = blockIdx.x * 128;
    const int n0 = blockIdx.y * 64;
    const float* W = (blockIdx.z == 0) ? Wq : (blockIdx.z == 1) ? Wk : Wv;
    bf16_t* Out = (blockIdx.z == 0) ? Qb : (blockIdx.z == 1) ? Kb : Vb;

#pragma unroll
    for (int e = 0; e < 32; ++e) {           // 128x64 f32 -> bf16
        int idx = tid + e * 256;
        ldsX[idx] = (bf16_t)x[(size_t)(m0 + (idx >> 6)) * 64 + (idx & 63)];
    }
#pragma unroll
    for (int e = 0; e < 16; ++e) {           // 64x64 slice of W
        int idx = tid + e * 256;
        ldsW[idx] = (bf16_t)W[(size_t)(idx >> 6) * HD + n0 + (idx & 63)];
    }
    __syncthreads();

    const bf16_t* arow = ldsX + (wave * 16 + (lane & 15)) * 64;
    v8f c[4];
#pragma unroll
    for (int nt = 0; nt < 4; ++nt) c[nt] = zero8();
#pragma unroll
    for (int kc = 0; kc < 2; ++kc) {
        v16bf a = load_a_frag(arow, kc, g);
#pragma unroll
        for (int nt = 0; nt < 4; ++nt)
            c[nt] = wmma_bf16(a, load_b_frag(ldsW, 64, kc, nt * 16, lane), c[nt]);
    }
    const int mBase = m0 + wave * 16 + 8 * g;
#pragma unroll
    for (int nt = 0; nt < 4; ++nt) {
        int col = n0 + nt * 16 + (lane & 15);
#pragma unroll
        for (int r = 0; r < 8; ++r)
            Out[(size_t)(mBase + r) * HD + col] = (bf16_t)c[nt][r];
    }
}

// ---------------------------------------------------------------- Q transpose
// Qb viewed (BH, T, 64) -> QT (BH, 64, T), bf16, padded LDS.
__global__ __launch_bounds__(256) void qt_kernel(const bf16_t* __restrict__ Qb,
                                                 bf16_t* __restrict__ QT) {
    __shared__ bf16_t tile[64 * 66];
    const int bh = blockIdx.y;
    const int j0 = blockIdx.x * 64;
    const int tid = threadIdx.x;
#pragma unroll
    for (int e = 0; e < 16; ++e) {
        int idx = tid + e * 256;
        int jr = idx >> 6, d = idx & 63;
        tile[jr * 66 + d] = Qb[((size_t)bh * T_SEQ + j0 + jr) * 64 + d];
    }
    __syncthreads();
#pragma unroll
    for (int e = 0; e < 16; ++e) {
        int idx = tid + e * 256;
        int d = idx >> 6, jr = idx & 63;
        QT[(size_t)bh * (64 * T_SEQ) + (size_t)d * T_SEQ + j0 + jr] = tile[jr * 66 + d];
    }
}

// ---------------------------------------------------------------- Flash attention
// Per workgroup: one head, 128 "query" rows (K-matrix rows), 8 waves x 16 rows.
// Streams j in 64-wide tiles (staged via async-to-LDS); S = K_i . Q_j^T via WMMA;
// online softmax on C-fragment row layout; acc += P@V.
__global__ __launch_bounds__(256) void attn_kernel(const bf16_t* __restrict__ Kb,
                                                   const bf16_t* __restrict__ QT,
                                                   const bf16_t* __restrict__ Vb,
                                                   bf16_t* __restrict__ Zb) {
    __shared__ __align__(16) bf16_t ldsQT[64 * 64];    // [d][j]
    __shared__ __align__(16) bf16_t ldsV[64 * 64];     // [j][d]
    __shared__ __align__(16) bf16_t ldsP[8][16 * 64];  // per-wave P tile [i][j]
    const int tid = threadIdx.x;
    const int lane = tid & 31, wave = tid >> 5;
    const int g = lane >> 4;
    const int bh = blockIdx.y;
    const int i0 = blockIdx.x * 128 + wave * 16;

    // Resident A fragments: this wave's 16 K-rows, pre-scaled by 1/sqrt(64).
    const bf16_t* krow = Kb + ((size_t)bh * T_SEQ + i0 + (lane & 15)) * 64;
    v16bf aK0 = load_a_frag(krow, 0, g);
    v16bf aK1 = load_a_frag(krow, 1, g);
#pragma unroll
    for (int i = 0; i < 16; ++i) {
        aK0[i] = (bf16_t)((float)aK0[i] * 0.125f);
        aK1[i] = (bf16_t)((float)aK1[i] * 0.125f);
    }

    v8f acc[4];
    float mrun[8], lrun[8];
#pragma unroll
    for (int dt = 0; dt < 4; ++dt) acc[dt] = zero8();
#pragma unroll
    for (int r = 0; r < 8; ++r) { mrun[r] = -3.0e38f; lrun[r] = 0.f; }

    const bf16_t* QTh = QT + (size_t)bh * (64 * T_SEQ);
    const bf16_t* Vh  = Vb + (size_t)bh * T_SEQ * 64;

    for (int jb = 0; jb < T_SEQ; jb += 64) {
        __syncthreads();   // all consumers done with previous tile
        // Async stage: QT tile [d][j] and V tile [j][d], 8 KB each,
        // 512 x 16B chunks each -> 2 chunks per thread per tile.
#pragma unroll
        for (int e = 0; e < 2; ++e) {
            int cidx = tid + e * 256;
            int row = cidx >> 3, sub = (cidx & 7) * 8;
            async_copy16(&ldsQT[row * 64 + sub], QTh + (size_t)row * T_SEQ + jb + sub);
            async_copy16(&ldsV[row * 64 + sub],  Vh + (size_t)(jb + row) * 64 + sub);
        }
        wait_async0();
        __syncthreads();

        // S tile (16 x 64) = (K_i/8) . Q_j^T
        v8f s[4];
#pragma unroll
        for (int jt = 0; jt < 4; ++jt) {
            v8f t = zero8();
            t = wmma_bf16(aK0, load_b_frag(ldsQT, 64, 0, jt * 16, lane), t);
            t = wmma_bf16(aK1, load_b_frag(ldsQT, 64, 1, jt * 16, lane), t);
            s[jt] = t;
        }

        // Row stats: C-fragment row m = r + 8*g lives across 16 lanes of this half-wave.
        float rmax[8];
#pragma unroll
        for (int r = 0; r < 8; ++r)
            rmax[r] = fmaxf(fmaxf(s[0][r], s[1][r]), fmaxf(s[2][r], s[3][r]));
#pragma unroll
        for (int off = 1; off < 16; off <<= 1)
#pragma unroll
            for (int r = 0; r < 8; ++r)
                rmax[r] = fmaxf(rmax[r], __shfl_xor(rmax[r], off, 32));

        float al[8];
#pragma unroll
        for (int r = 0; r < 8; ++r) {
            float mn = fmaxf(mrun[r], rmax[r]);
            al[r] = __expf(mrun[r] - mn);
            mrun[r] = mn;
        }
        float psum[8];
#pragma unroll
        for (int r = 0; r < 8; ++r) psum[r] = 0.f;
#pragma unroll
        for (int jt = 0; jt < 4; ++jt)
#pragma unroll
            for (int r = 0; r < 8; ++r) {
                float p = __expf(s[jt][r] - mrun[r]);
                s[jt][r] = p;
                psum[r] += p;
            }
#pragma unroll
        for (int off = 1; off < 16; off <<= 1)
#pragma unroll
            for (int r = 0; r < 8; ++r)
                psum[r] += __shfl_xor(psum[r], off, 32);
#pragma unroll
        for (int r = 0; r < 8; ++r) lrun[r] = lrun[r] * al[r] + psum[r];
#pragma unroll
        for (int dt = 0; dt < 4; ++dt)
#pragma unroll
            for (int r = 0; r < 8; ++r) acc[dt][r] *= al[r];

        // P (bf16) to per-wave LDS, then re-read in A-fragment order.
        bf16_t* P = ldsP[wave];
#pragma unroll
        for (int jt = 0; jt < 4; ++jt)
#pragma unroll
            for (int r = 0; r < 8; ++r)
                P[(r + 8 * g) * 64 + jt * 16 + (lane & 15)] = (bf16_t)s[jt][r];
        asm volatile("s_wait_dscnt 0" ::: "memory");

        const bf16_t* prow = P + (lane & 15) * 64;
#pragma unroll
        for (int kc = 0; kc < 2; ++kc) {
            v16bf aP = load_a_frag(prow, kc, g);
#pragma unroll
            for (int dt = 0; dt < 4; ++dt)
                acc[dt] = wmma_bf16(aP, load_b_frag(ldsV, 64, kc, dt * 16, lane), acc[dt]);
        }
    }

    // normalize and store Z (bf16)
#pragma unroll
    for (int dt = 0; dt < 4; ++dt) {
        int col = dt * 16 + (lane & 15);
#pragma unroll
        for (int r = 0; r < 8; ++r) {
            int m = i0 + r + 8 * g;
            Zb[((size_t)bh * T_SEQ + m) * 64 + col] = (bf16_t)(acc[dt][r] / lrun[r]);
        }
    }
}

// ---------------------------------------------------------------- Output GEMM
// (8192x768 bf16) @ (768x64 f32->bf16) -> f32 out, K streamed in 64-chunks.
__global__ __launch_bounds__(256) void out_kernel(const bf16_t* __restrict__ Zb,
                                                  const float* __restrict__ Wo,
                                                  float* __restrict__ out) {
    __shared__ __align__(16) bf16_t ldsZ[128 * 64];
    __shared__ __align__(16) bf16_t ldsW[64 * 64];
    const int tid = threadIdx.x;
    const int lane = tid & 31, wave = tid >> 5;
    const int g = lane >> 4;
    const int m0 = blockIdx.x * 128;

    v8f c[4];
#pragma unroll
    for (int nt = 0; nt < 4; ++nt) c[nt] = zero8();

    for (int k0 = 0; k0 < HD; k0 += 64) {
        __syncthreads();
        // Z tile (128x64 bf16, 16 KB) via async-to-LDS: 1024 x 16B chunks.
#pragma unroll
        for (int e = 0; e < 4; ++e) {
            int cidx = tid + e * 256;
            int row = cidx >> 3, sub = (cidx & 7) * 8;
            async_copy16(&ldsZ[row * 64 + sub],
                         Zb + (size_t)(m0 + row) * HD + k0 + sub);
        }
        // Wo chunk needs f32->bf16 conversion: manual stage.
#pragma unroll
        for (int e = 0; e < 16; ++e) {
            int idx = tid + e * 256;
            ldsW[idx] = (bf16_t)Wo[(size_t)(k0 + (idx >> 6)) * 64 + (idx & 63)];
        }
        wait_async0();
        __syncthreads();

        const bf16_t* arow = ldsZ + (wave * 16 + (lane & 15)) * 64;
#pragma unroll
        for (int kc = 0; kc < 2; ++kc) {
            v16bf a = load_a_frag(arow, kc, g);
#pragma unroll
            for (int nt = 0; nt < 4; ++nt)
                c[nt] = wmma_bf16(a, load_b_frag(ldsW, 64, kc, nt * 16, lane), c[nt]);
        }
    }
    const int mBase = m0 + wave * 16 + 8 * g;
#pragma unroll
    for (int nt = 0; nt < 4; ++nt) {
        int col = nt * 16 + (lane & 15);
#pragma unroll
        for (int r = 0; r < 8; ++r)
            out[(size_t)(mBase + r) * 64 + col] = c[nt][r];
    }
}

// ----------------------------------------------------------------
extern "C" void kernel_launch(void* const* d_in, const int* in_sizes, int n_in,
                              void* d_out, int out_size, void* d_ws, size_t ws_size,
                              hipStream_t stream) {
    (void)in_sizes; (void)n_in; (void)out_size; (void)ws_size;
    const float* x  = (const float*)d_in[0];
    const float* Wq = (const float*)d_in[1];
    const float* Wk = (const float*)d_in[2];
    const float* Wv = (const float*)d_in[3];
    const float* Wo = (const float*)d_in[4];
    float* out = (float*)d_out;

    const size_t SEG = (size_t)ROWS * HD * sizeof(bf16_t); // 12.58 MB per buffer
    char* ws = (char*)d_ws;
    bf16_t* Qb = (bf16_t*)(ws);
    bf16_t* Kb = (bf16_t*)(ws + SEG);
    bf16_t* Vb = (bf16_t*)(ws + 2 * SEG);
    bf16_t* QT = (bf16_t*)(ws + 3 * SEG);
    bf16_t* Zb = Qb;  // Q region is dead after the transpose

    qkv_kernel<<<dim3(ROWS / 128, HD / 64, 3), 256, 0, stream>>>(x, Wq, Wk, Wv, Qb, Kb, Vb);
    qt_kernel<<<dim3(T_SEQ / 64, BH), 256, 0, stream>>>(Qb, QT);
    attn_kernel<<<dim3(T_SEQ / 128, BH), 256, 0, stream>>>(Kb, QT, Vb, Zb);
    out_kernel<<<ROWS / 128, 256, 0, stream>>>(Zb, Wo, out);
}
